// FixedAdaptiveEdgeConvBlock_87033217286473
// MI455X (gfx1250) — matrix operand order repaired and checked by reading, not compiled, not run on hardware
//
#include <hip/hip_runtime.h>
#include <hip/hip_bf16.h>

typedef __attribute__((ext_vector_type(16))) _Float16 v16h;
typedef __attribute__((ext_vector_type(8)))  float    v8f;

#define N_PTS 8192
#define C_IN  128
#define D_HID 256
#define KNN   16
#define LDS_STRIDE 264   // halves per row: 256 + 8 pad (528B = 132 dwords, bank-skewed)

union Frag { v16h h; uint4 q[2]; };

// ---------------------------------------------------------------------------
// Kernel 0: convert W1 (2C x D) and W2 (D x D) from f32 row-major [k][n]
// to f16 transposed [n][k] so WMMA B fragments are K-contiguous 32B loads.
// grid = 512 blocks x 256 threads.
// ---------------------------------------------------------------------------
__global__ void convert_weights(const float* __restrict__ W1,
                                const float* __restrict__ W2,
                                _Float16* __restrict__ W1T,
                                _Float16* __restrict__ W2T) {
    int b = blockIdx.x;
    int k = threadIdx.x;
    if (b < D_HID) {
        W1T[b * 256 + k] = (_Float16)W1[k * 256 + b];
    } else {
        int n = b - D_HID;
        W2T[n * 256 + k] = (_Float16)W2[k * 256 + n];
    }
}

// ---------------------------------------------------------------------------
// Kernel 1: exact kNN (k=16, excluding self). One query per thread,
// candidates tiled through LDS, register-resident ascending top-16 list.
// grid = 64 blocks x 128 threads.
// ---------------------------------------------------------------------------
__global__ void knn_kernel(const float* __restrict__ pos,
                           int* __restrict__ idx_out) {
    __shared__ float tile[128 * 3];
    const int q = blockIdx.x * 128 + threadIdx.x;
    const float qx = pos[q * 3 + 0];
    const float qy = pos[q * 3 + 1];
    const float qz = pos[q * 3 + 2];

    float bd[KNN];
    int   bi[KNN];
#pragma unroll
    for (int s = 0; s < KNN; ++s) { bd[s] = 3.4e38f; bi[s] = 0; }

    for (int t = 0; t < N_PTS / 128; ++t) {
        for (int i = threadIdx.x; i < 128 * 3; i += 128)
            tile[i] = pos[t * 128 * 3 + i];
        __syncthreads();

        for (int j = 0; j < 128; ++j) {
            const int jg = t * 128 + j;
            const float dx = tile[j * 3 + 0] - qx;
            const float dy = tile[j * 3 + 1] - qy;
            const float dz = tile[j * 3 + 2] - qz;
            const float d  = dx * dx + dy * dy + dz * dz;
            if (jg != q && d < bd[KNN - 1]) {
                float cd = d; int ci = jg;
#pragma unroll
                for (int s = 0; s < KNN; ++s) {
                    const bool c = cd < bd[s];
                    const float td = bd[s]; const int ti = bi[s];
                    bd[s] = c ? cd : td;  bi[s] = c ? ci : ti;
                    cd    = c ? td : cd;  ci    = c ? ti : ci;
                }
            }
        }
        __syncthreads();
    }
#pragma unroll
    for (int s = 0; s < KNN; ++s) idx_out[q * KNN + s] = bi[s];
}

// ---------------------------------------------------------------------------
// WMMA GEMM over one 64x256 f16 A-tile in LDS against a 256x256 f16 B
// (stored transposed [n][k] in global). Each wave computes 64 rows x 32 cols.
// A fragment layout (16-bit A 16x32): lanes 0-15 row M=lane, K=0-7 & 16-23;
// lanes 16-31 same rows, K=8-15 & 24-31  -> two 16B LDS loads per tile.
// B fragment: column (nbase+nt*16+lane&15), same K order as A:
// lanes 0-15 K=0-7 & 16-23, lanes 16-31 K=8-15 & 24-31 -> two 16B global loads.
// ---------------------------------------------------------------------------
__device__ __forceinline__ void wave_gemm(const _Float16* __restrict__ smem,
                                          const _Float16* __restrict__ WT,
                                          int lane, int nbase, v8f acc[4][2]) {
    const int row16 = lane & 15;
    const int hi8   = (lane >> 4) * 8;
    const int khalf = (lane >> 4) * 16;
    for (int kt = 0; kt < 8; ++kt) {
        Frag a[4];
#pragma unroll
        for (int mt = 0; mt < 4; ++mt) {
            const _Float16* rp = smem + (mt * 16 + row16) * LDS_STRIDE + kt * 32 + hi8;
            a[mt].q[0] = *(const uint4*)(rp);
            a[mt].q[1] = *(const uint4*)(rp + 16);
        }
        Frag b[2];
#pragma unroll
        for (int nt = 0; nt < 2; ++nt) {
            const _Float16* wp = WT + (nbase + nt * 16 + row16) * 256 + kt * 32 + hi8;
            b[nt].q[0] = *(const uint4*)(wp);
            b[nt].q[1] = *(const uint4*)(wp + 16);
        }
#pragma unroll
        for (int mt = 0; mt < 4; ++mt)
#pragma unroll
            for (int nt = 0; nt < 2; ++nt)
                acc[mt][nt] = __builtin_amdgcn_wmma_f32_16x16x32_f16(
                    false, a[mt].h, false, b[nt].h,
                    (short)0, acc[mt][nt], false, false);
    }
}

// ---------------------------------------------------------------------------
// Kernel 2: fused EdgeConv MLP. Block = 256 threads (8 waves) = 4 points
// (64 edge rows). E = [x_i | x_j - x_i] staged as f16 in LDS, GEMM1 + ReLU,
// h1 repacked into the same LDS buffer, GEMM2, per-point max over 16 rows.
// grid = 2048 blocks x 256 threads.
// ---------------------------------------------------------------------------
__global__ void edgeconv_kernel(const float* __restrict__ x,
                                const int* __restrict__ nidx,
                                const _Float16* __restrict__ W1T,
                                const float* __restrict__ b1,
                                const _Float16* __restrict__ W2T,
                                const float* __restrict__ b2,
                                float* __restrict__ out) {
    __shared__ __align__(16) _Float16 smem[64 * LDS_STRIDE];  // 33 KB: E, then h1
    const int tid  = threadIdx.x;
    const int lane = tid & 31;
    const int wave = tid >> 5;
    const int p0   = blockIdx.x * 4;
    const int nbase = wave * 32;                 // this wave's output columns

    // --- Stage E tile: 64 rows x 256 cols f16 ---
    for (int it = tid; it < 64 * C_IN; it += 256) {
        const int row = it >> 7;                 // 0..63
        const int c   = it & (C_IN - 1);
        const int pt  = p0 + (row >> 4);
        const int nb  = row & 15;
        const int j   = nidx[pt * KNN + nb];
        const float xi = x[pt * C_IN + c];
        const float xj = x[j  * C_IN + c];
        smem[row * LDS_STRIDE + c]        = (_Float16)xi;
        smem[row * LDS_STRIDE + C_IN + c] = (_Float16)(xj - xi);
    }
    __syncthreads();

    const int col16 = lane & 15;
    const int hi    = lane >> 4;

    v8f acc[4][2];
#pragma unroll
    for (int mt = 0; mt < 4; ++mt)
#pragma unroll
        for (int nt = 0; nt < 2; ++nt)
            acc[mt][nt] = (v8f){0.f, 0.f, 0.f, 0.f, 0.f, 0.f, 0.f, 0.f};

    // --- GEMM1: h1 = relu(E @ W1 + b1) ---
    wave_gemm(smem, W1T, lane, nbase, acc);
    __syncthreads();                             // all waves done reading E

#pragma unroll
    for (int nt = 0; nt < 2; ++nt) {
        const int colg = nbase + nt * 16 + col16;
        const float bias = b1[colg];
#pragma unroll
        for (int mt = 0; mt < 4; ++mt)
#pragma unroll
            for (int r = 0; r < 8; ++r) {
                float v = acc[mt][nt][r] + bias;
                v = v > 0.f ? v : 0.f;
                const int row = mt * 16 + hi * 8 + r;
                smem[row * LDS_STRIDE + colg] = (_Float16)v;
            }
    }
    __syncthreads();                             // h1 tile fully in LDS

#pragma unroll
    for (int mt = 0; mt < 4; ++mt)
#pragma unroll
        for (int nt = 0; nt < 2; ++nt)
            acc[mt][nt] = (v8f){0.f, 0.f, 0.f, 0.f, 0.f, 0.f, 0.f, 0.f};

    // --- GEMM2: h2 = h1 @ W2 + b2 ---
    wave_gemm(smem, W2T, lane, nbase, acc);

    // --- Segmented max over each point's 16 edge rows (one M-tile/point) ---
#pragma unroll
    for (int nt = 0; nt < 2; ++nt) {
        const int colg = nbase + nt * 16 + col16;
        const float bias = b2[colg];
#pragma unroll
        for (int mt = 0; mt < 4; ++mt) {
            float m = -3.4e38f;
#pragma unroll
            for (int r = 0; r < 8; ++r) {
                const float v = acc[mt][nt][r] + bias;
                m = v > m ? v : m;
            }
            const float o = __shfl_xor(m, 16, 32);  // rows 0-7 vs 8-15
            m = m > o ? m : o;
            if (lane < 16)
                out[(p0 + mt) * D_HID + colg] = m;
        }
    }
}

// ---------------------------------------------------------------------------
extern "C" void kernel_launch(void* const* d_in, const int* in_sizes, int n_in,
                              void* d_out, int out_size, void* d_ws, size_t ws_size,
                              hipStream_t stream) {
    const float* x   = (const float*)d_in[0];   // (8192, 128)
    const float* pos = (const float*)d_in[1];   // (8192, 3)
    const float* W1  = (const float*)d_in[2];   // (256, 256)
    const float* b1  = (const float*)d_in[3];   // (256,)
    const float* W2  = (const float*)d_in[4];   // (256, 256)
    const float* b2  = (const float*)d_in[5];   // (256,)
    float* out = (float*)d_out;                 // (8192, 256)

    char* ws = (char*)d_ws;
    _Float16* W1T   = (_Float16*)(ws);                 // 128 KB
    _Float16* W2T   = (_Float16*)(ws + 131072);        // 128 KB
    int*      idxbf = (int*)     (ws + 262144);        // 512 KB

    convert_weights<<<512, 256, 0, stream>>>(W1, W2, W1T, W2T);
    knn_kernel<<<N_PTS / 128, 128, 0, stream>>>(pos, idxbf);
    edgeconv_kernel<<<N_PTS / 4, 256, 0, stream>>>(x, idxbf, W1T, b1, W2T, b2, out);
}
